// CorefScore_29661044146891
// MI455X (gfx1250) — matrix operand order repaired and checked by reading, not compiled, not run on hardware
//
#include <hip/hip_runtime.h>
#include <hip/hip_bf16.h>

// CorefScore on gfx1250 (MI455X). Dominant cost: bilinear pair term
// C[m,k,h] = sum_d x_m[d]*x_{m-K+k}[d]*W1c[d,h]  (~27.6 GFLOP) -> f16 WMMA
// (V_WMMA_F32_16X16X32_F16) with f32 accumulation. Everything fits in L2.

typedef __attribute__((ext_vector_type(16))) _Float16 v16h;
typedef __attribute__((ext_vector_type(8)))  _Float16 v8h;
typedef __attribute__((ext_vector_type(8)))  float    v8f;

#define MM   2048
#define DD   900
#define HH   150
#define KK   50          // K_DEF (device scalar not host-readable under graph capture)
#define DP   1024        // depth padded to 32*32 WMMA steps
#define YS   480         // Y row stride: 3 blocks of 160 cols ([w1m | W1a | W1b])
#define OUTW (KK + 1)
#define NEGV (-1.0e9f)

// workspace layout (bytes); total ~9.5 MB
#define OFF_XF16   0u                         // 2048*1024*2 = 4,194,304
#define OFF_W1CT   (4u*1024u*1024u)           //  160*1024*2 =   327,680
#define OFF_WALLT  (OFF_W1CT + 512u*1024u)    //  480*1024*2 =   983,040
#define OFF_Y      (OFF_WALLT + 1024u*1024u)  // 2048*480*4  = 3,932,160
#define OFF_MENT   (OFF_Y + 4u*1024u*1024u)   // 2048*4

// ---------------------------------------------------------------- prep ------
// Xf:    [2048][1024] f16, cols >=900 zero
// W1cT:  [160][1024]  f16, W1cT[n][d] = w1p[1800+d][n]   (h-major for B frags)
// WallT: [480][1024]  f16, blocks of 160 rows: w1m^T | W1a^T | W1b^T
__global__ void prep_kernel(const float* __restrict__ X,
                            const float* __restrict__ w1m,
                            const float* __restrict__ w1p,
                            _Float16* __restrict__ Xf,
                            _Float16* __restrict__ W1cT,
                            _Float16* __restrict__ WallT) {
    const int n_x = MM * DP, n_c = 160 * DP, n_w = 480 * DP;
    const int total = n_x + n_c + n_w;
    for (int i = blockIdx.x * blockDim.x + threadIdx.x; i < total;
         i += gridDim.x * blockDim.x) {
        if (i < n_x) {
            int r = i >> 10, d = i & (DP - 1);
            Xf[i] = (d < DD) ? (_Float16)X[r * DD + d] : (_Float16)0.f;
        } else if (i < n_x + n_c) {
            int j = i - n_x, n = j >> 10, d = j & (DP - 1);
            W1cT[j] = (n < HH && d < DD) ? (_Float16)w1p[(2 * DD + d) * HH + n]
                                         : (_Float16)0.f;
        } else {
            int j = i - n_x - n_c, n = j >> 10, d = j & (DP - 1);
            int blk = n / 160, nn = n - blk * 160;
            float v = 0.f;
            if (nn < HH && d < DD) {
                if (blk == 0)      v = w1m[d * HH + nn];
                else if (blk == 1) v = w1p[d * HH + nn];
                else               v = w1p[(DD + d) * HH + nn];
            }
            WallT[j] = (_Float16)v;
        }
    }
}

// -------------------------------------------------------------- helpers -----
__device__ inline v16h make_a16(v8h lo, v8h hi) {
    return __builtin_shufflevector(lo, hi, 0, 1, 2, 3, 4, 5, 6, 7,
                                           8, 9, 10, 11, 12, 13, 14, 15);
}

// ------------------------------------------------------------- rowgemm -----
// Y[2048][480] = Xf @ WallT^T : 64 rows x 160 cols per WG, 10 waves,
// wave w owns col tile n0 = by*160 + 16w, four 16-row tiles.
__global__ __launch_bounds__(320)
void rowgemm_kernel(const _Float16* __restrict__ Xf,
                    const _Float16* __restrict__ WallT,
                    float* __restrict__ Y) {
    const int lane = threadIdx.x & 31;
    const int wid  = threadIdx.x >> 5;
    const int m0   = blockIdx.x * 64;
    const int n0   = blockIdx.y * 160 + wid * 16;
    const int l15  = lane & 15;
    const int hi   = lane >> 4;   // 0: lanes 0-15, 1: lanes 16-31

    v8f acc[4] = {};
    for (int d0 = 0; d0 < DP; d0 += 32) {
        // B frag: lane -> col n0+l15; halfs = K d0+hi*16 .. +15 (contiguous)
        const _Float16* bp = WallT + (size_t)(n0 + l15) * DP + d0 + hi * 16;
        v16h bfrag = *(const v16h*)bp;
#pragma unroll
        for (int t = 0; t < 4; ++t) {
            // A frag: lane -> row m0+16t+l15; halfs = K d0+hi*16 .. +15 (contiguous)
            const _Float16* ap =
                Xf + (size_t)(m0 + t * 16 + l15) * DP + d0 + hi * 16;
            v8h alo = *(const v8h*)ap;
            v8h ahi = *(const v8h*)(ap + 8);
            v16h afrag = make_a16(alo, ahi);
            acc[t] = __builtin_amdgcn_wmma_f32_16x16x32_f16(
                false, afrag, false, bfrag, (short)0, acc[t], false, false);
        }
    }
#pragma unroll
    for (int t = 0; t < 4; ++t)
#pragma unroll
        for (int v = 0; v < 8; ++v) {
            int row = m0 + t * 16 + v + hi * 8;
            Y[(size_t)row * YS + n0 + l15] = acc[t][v];
        }
}

// --------------------------------------------------------------- mentk -----
// ment[m] = ReLU(Y[m, 0:150] + b1m) . w2m + b2m
__global__ __launch_bounds__(256)
void ment_kernel(const float* __restrict__ Y, const float* __restrict__ b1m,
                 const float* __restrict__ w2m, const float* __restrict__ b2m,
                 float* __restrict__ ment) {
    const int lane = threadIdx.x & 31;
    const int row  = blockIdx.x * 8 + (threadIdx.x >> 5);
    float s = 0.f;
    for (int h = lane; h < HH; h += 32)
        s += fmaxf(Y[(size_t)row * YS + h] + b1m[h], 0.f) * w2m[h];
#pragma unroll
    for (int off = 16; off; off >>= 1) s += __shfl_xor(s, off, 32);
    if (lane == 0) ment[row] = s + b2m[0];
}

// --------------------------------------------------------------- pairk -----
#define CH  128   // LDS chunk depth (halfs) = 4 WMMA K-steps
#define PST 136   // LDS row stride in halfs (136h = 68 DW; 68%64=4 -> conflict-free b128)

__global__ __launch_bounds__(320)
void pair_kernel(const _Float16* __restrict__ Xf,
                 const _Float16* __restrict__ W1cT,
                 const float* __restrict__ Y,
                 const float* __restrict__ ment,
                 const float* __restrict__ b1p,
                 const float* __restrict__ w2p,
                 const float* __restrict__ b2p,
                 float* __restrict__ out) {
    __shared__ _Float16 P[64 * PST];   // ~17.4 KB product tile
    __shared__ float pacc[64];

    const int m    = blockIdx.x;
    const int tid  = threadIdx.x;
    const int lane = tid & 31;
    const int wid  = tid >> 5;     // 0..9 -> h tile
    const int l15  = lane & 15;
    const int hi   = lane >> 4;
    const int n0   = wid * 16;

    if (tid < 64) pacc[tid] = 0.f;

    v8f acc[4] = {};
    for (int c = 0; c < DP / CH; ++c) {
        __syncthreads();
        // Build P[k][dl] = Xf[m][d] * Xf[m-50+k][d] for this 128-deep chunk.
        for (int task = tid; task < 64 * (CH / 8); task += 320) {
            int k  = task >> 4;
            int dl = (task & 15) * 8;
            int d  = c * CH + dl;
            int j  = m - KK + k;
            v8h p = {};
            if (k < KK && j >= 0) {
                v8h a = *(const v8h*)(Xf + (size_t)m * DP + d);
                v8h b = *(const v8h*)(Xf + (size_t)j * DP + d);
                p = a * b;
            }
            *(v8h*)(P + k * PST + dl) = p;
        }
        __syncthreads();
        // Consume: 4 K-steps x 4 row tiles of WMMA against W1c column tile.
        for (int s = 0; s < 4; ++s) {
            int dg = c * CH + s * 32;
            const _Float16* bp = W1cT + (size_t)(n0 + l15) * DP + dg + hi * 16;
            v16h bfrag = *(const v16h*)bp;
#pragma unroll
            for (int t = 0; t < 4; ++t) {
                const _Float16* ap = P + (t * 16 + l15) * PST + s * 32 + hi * 16;
                v8h alo = *(const v8h*)ap;
                v8h ahi = *(const v8h*)(ap + 8);
                v16h afrag = make_a16(alo, ahi);
                acc[t] = __builtin_amdgcn_wmma_f32_16x16x32_f16(
                    false, afrag, false, bfrag, (short)0, acc[t], false, false);
            }
        }
    }

    // Epilogue: act = ReLU(U[m]+V[j]+C+b1p); pair = act . w2p
    const int  hloc = n0 + l15;
    const bool hok  = hloc < HH;
    const float um = hok ? Y[(size_t)m * YS + 160 + hloc] : 0.f;
    const float b1 = hok ? b1p[hloc] : 0.f;
    const float w2 = hok ? w2p[hloc] : 0.f;
#pragma unroll
    for (int t = 0; t < 4; ++t) {
#pragma unroll
        for (int v = 0; v < 8; ++v) {
            int k = t * 16 + v + hi * 8;
            int j = m - KK + k;
            float contrib = 0.f;
            if (k < KK && j >= 0 && hok) {
                float vv = Y[(size_t)j * YS + 320 + hloc];
                float a  = fmaxf(um + vv + acc[t][v] + b1, 0.f);
                contrib  = a * w2;
            }
            // sum over the 16 h-lanes of each half-wave
#pragma unroll
            for (int off = 8; off; off >>= 1)
                contrib += __shfl_xor(contrib, off, 32);
            if (l15 == 0 && k < 64) atomicAdd(&pacc[k], contrib);
        }
    }
    __syncthreads();
    if (tid <= KK) {
        float r;
        if (tid == KK) {
            r = 0.f;                                    // dummy column
        } else {
            int j = m - KK + tid;
            r = (j >= 0) ? (ment[m] + ment[j] + pacc[tid] + b2p[0]) : NEGV;
        }
        out[(size_t)m * OUTW + tid] = r;
    }
}

// -------------------------------------------------------------- launch -----
extern "C" void kernel_launch(void* const* d_in, const int* in_sizes, int n_in,
                              void* d_out, int out_size, void* d_ws,
                              size_t ws_size, hipStream_t stream) {
    const float* X   = (const float*)d_in[0];
    const float* w1m = (const float*)d_in[1];
    const float* b1m = (const float*)d_in[2];
    const float* w2m = (const float*)d_in[3];
    const float* b2m = (const float*)d_in[4];
    const float* w1p = (const float*)d_in[5];
    const float* b1p = (const float*)d_in[6];
    const float* w2p = (const float*)d_in[7];
    const float* b2p = (const float*)d_in[8];
    float* out = (float*)d_out;

    char* ws = (char*)d_ws;
    _Float16* Xf    = (_Float16*)(ws + OFF_XF16);
    _Float16* W1cT  = (_Float16*)(ws + OFF_W1CT);
    _Float16* WallT = (_Float16*)(ws + OFF_WALLT);
    float*    Y     = (float*)(ws + OFF_Y);
    float*    ment  = (float*)(ws + OFF_MENT);

    prep_kernel<<<1024, 256, 0, stream>>>(X, w1m, w1p, Xf, W1cT, WallT);
    dim3 g1(32, 3);
    rowgemm_kernel<<<g1, 320, 0, stream>>>(Xf, WallT, Y);
    ment_kernel<<<256, 256, 0, stream>>>(Y, b1m, w2m, b2m, ment);
    pair_kernel<<<2048, 320, 0, stream>>>(Xf, W1cT, Y, ment, b1p, w2p, b2p, out);
}
